// GCBlock_42039139893971
// MI455X (gfx1250) — matrix operand order repaired and checked by reading, not compiled, not run on hardware
//
#include <hip/hip_runtime.h>
#include <hip/hip_bf16.h>

// ---------------------------------------------------------------------------
// Problem constants (from the reference)
// ---------------------------------------------------------------------------
static constexpr int NN = 10000;    // nodes
static constexpr int EE = 320000;   // edges
static constexpr int CC = 64;       // channels

// d_out layout (floats): p1_out | i1 | p3 | i3
static constexpr long O_P1 = 0;
static constexpr long O_I1 = (long)NN * CC;                  // 640,000
static constexpr long O_P3 = O_I1 + (long)EE * CC;           // 21,120,000
static constexpr long O_I3 = O_P3 + (long)NN * 3 * CC;       // 23,040,000

// d_ws layout (bytes)
static constexpr size_t WS_P_OFF  = 0;                       // p  : N*64 f32
static constexpr size_t WS_PX_OFF = (size_t)NN * CC * 4;     // px : N*64 f32
static constexpr size_t WS_W_OFF  = WS_PX_OFF * 2;           // swizzled f16 weights
// swizzled weight offsets (in _Float16 elements from WS_W_OFF)
static constexpr int W_PPW0 = 0;
static constexpr int W_PPW1 = 4096;
static constexpr int W_PIW0 = 8192;
static constexpr int W_PIW1 = 12288;      // 64x512 = 32768 halves
static constexpr int W_IIW0 = 45056;
static constexpr int W_IIW1 = 49152;
static constexpr int W_EPP  = 53248;
static constexpr int W_EWI  = 57344;
static constexpr int W_EWJ  = 61440;
static constexpr int W_EII  = 65536;

static constexpr int WAVES = 4;           // waves per block (128 threads)

typedef __attribute__((ext_vector_type(16))) _Float16 v16h;
typedef __attribute__((ext_vector_type(8)))  _Float16 v8h;
typedef __attribute__((ext_vector_type(2)))  __fp16   v2fp;   // cvt_pkrtz result type
typedef __attribute__((ext_vector_type(8)))  float    v8f;

// ---------------------------------------------------------------------------
// Fast helpers
// ---------------------------------------------------------------------------
// Branch-free tanh on the HW transcendental pipe:
// tanh(x) = 1 - 2 / (e^(2x) + 1) ; e^(2x) = exp2(2*log2(e)*x)
// Saturates correctly: x->+inf => 1, x->-inf => -1. No EXEC divergence.
__device__ __forceinline__ float fast_tanh(float x) {
  float e = __builtin_amdgcn_exp2f(x * 2.88539008177792681472f);
  return 1.0f - 2.0f * __builtin_amdgcn_rcpf(e + 1.0f);
}

// Packed f32x2 -> f16x2 store into an LDS half buffer (v_cvt_pk_rtz_f16_f32).
__device__ __forceinline__ void st_pk16(_Float16* dst, float a, float b) {
  *(v2fp*)dst = __builtin_amdgcn_cvt_pkrtz(a, b);
}

// ---------------------------------------------------------------------------
// WMMA helpers
// ---------------------------------------------------------------------------
__device__ __forceinline__ v8f wmma16(v16h a, v16h b, v8f c) {
  return __builtin_amdgcn_wmma_f32_16x16x32_f16(
      /*neg_a=*/false, a, /*neg_b=*/false, b,
      /*c_mod=*/(short)0, c, /*reuse_a=*/false, /*reuse_b=*/false);
}

// A fragment (16x32 f16) from an LDS row-major 16x64 f16 tile.
// Lane L: m = L&15, elems 0..7 -> K = kbase + hi8 + i ; 8..15 -> K = kbase+16+hi8+i
__device__ __forceinline__ v16h ldsA(const _Float16* row, int kbase, int hi8) {
  v8h a0 = *(const v8h*)(row + kbase + hi8);
  v8h a1 = *(const v8h*)(row + kbase + 16 + hi8);
  v16h a;
#pragma unroll
  for (int i = 0; i < 8; ++i) { a[i] = a0[i]; a[i + 8] = a1[i]; }
  return a;
}

// B fragment from pre-swizzled weights: 512 halves per 32x16 tile, 16/lane.
__device__ __forceinline__ v16h ldB(const _Float16* __restrict__ W, int tile, int lane) {
  return *(const v16h*)(W + (size_t)tile * 512 + lane * 16);
}

// Y(16x64,LDS f16) = [tanh]( X(16x64,LDS f16) @ W(64x64 swizzled) )
__device__ __forceinline__ void stage64(const _Float16* Xl, const _Float16* __restrict__ W,
                                        _Float16* Yl, int lane, bool dotanh) {
  const int m = lane & 15, mb = (lane >> 4) * 8, hi8 = mb;
  v16h a0 = ldsA(Xl + m * 64, 0, hi8);
  v16h a1 = ldsA(Xl + m * 64, 32, hi8);
#pragma unroll
  for (int nt = 0; nt < 4; ++nt) {
    v8f c = {};
    c = wmma16(a0, ldB(W, nt, lane), c);
    c = wmma16(a1, ldB(W, 4 + nt, lane), c);
    const int n = nt * 16 + m;
#pragma unroll
    for (int r = 0; r < 8; ++r) {
      float v = c[r];
      Yl[(mb + r) * 64 + n] = (_Float16)(dotanh ? fast_tanh(v) : v);
    }
  }
}

// gout rows [rowBase..rowBase+15], 64 cols f32 = X(16x64,LDS f16) @ W(64x64)
__device__ __forceinline__ void stageOut(const _Float16* Xl, const _Float16* __restrict__ W,
                                         float* __restrict__ gout, long rowBase, int lane) {
  const int m = lane & 15, mb = (lane >> 4) * 8, hi8 = mb;
  v16h a0 = ldsA(Xl + m * 64, 0, hi8);
  v16h a1 = ldsA(Xl + m * 64, 32, hi8);
#pragma unroll
  for (int nt = 0; nt < 4; ++nt) {
    v8f c = {};
    c = wmma16(a0, ldB(W, nt, lane), c);
    c = wmma16(a1, ldB(W, 4 + nt, lane), c);
    const int n = nt * 16 + m;
#pragma unroll
    for (int r = 0; r < 8; ++r) gout[(rowBase + mb + r) * 64 + n] = c[r];
  }
}

// ---------------------------------------------------------------------------
// Kernel: f32 weight (K x Nw, row-major) -> f16 swizzled B-fragment layout
// ---------------------------------------------------------------------------
__global__ void k_swz(const float* __restrict__ src, _Float16* __restrict__ dst,
                      int K, int Nw) {
  int t = blockIdx.x * blockDim.x + threadIdx.x;
  if (t >= K * Nw) return;
  int tile = t >> 9, within = t & 511;
  int lane = within >> 4, i = within & 15;
  int ntiles = Nw >> 4;
  int kt = tile / ntiles, nt = tile - kt * ntiles;
  int n = (nt << 4) + (lane & 15);
  int k = (kt << 5) + ((i >> 3) << 4) + ((lane >> 4) << 3) + (i & 7);
  dst[t] = (_Float16)src[k * Nw + n];
}

// Zero the segment-sum regions of d_out (p1_out and p3).
__global__ void k_zero(float* __restrict__ out) {
  long i = (long)blockIdx.x * blockDim.x + threadIdx.x;
  if (i < (long)NN * CC) out[O_P1 + i] = 0.f;
  else if (i < (long)NN * 256) out[O_P3 + (i - (long)NN * CC)] = 0.f;
}

// ---------------------------------------------------------------------------
// Kernel A: p = tanh(p1 @ pp_w0) @ pp_w1   (16 nodes per wave)
// ---------------------------------------------------------------------------
__global__ void k_node_mlp(const float* __restrict__ p1, const _Float16* __restrict__ w0,
                           const _Float16* __restrict__ w1, float* __restrict__ pout) {
  __shared__ __align__(16) _Float16 sX[WAVES][1024];
  __shared__ __align__(16) _Float16 sH[WAVES][1024];
  const int lane = threadIdx.x & 31, wv = threadIdx.x >> 5;
  int tile = blockIdx.x * WAVES + wv;
  const int tmax = NN / 16 - 1;
  if (tile > tmax) tile = tmax;                  // clamp: duplicates write same data
  const long nb = (long)tile * 16;
  _Float16* X = sX[wv];
#pragma unroll
  for (int r = 0; r < 16; ++r) {
    float2 v = *(const float2*)(p1 + (nb + r) * 64 + 2 * lane);
    st_pk16(X + r * 64 + 2 * lane, v.x, v.y);
  }
  __syncthreads();
  stage64(X, w0, sH[wv], lane, /*tanh=*/true);
  __syncthreads();
  stageOut(sH[wv], w1, pout, nb, lane);
}

// ---------------------------------------------------------------------------
// Kernel C: y = x @ W (no activation), 16 nodes per wave (px = p1_out @ e_pp)
// ---------------------------------------------------------------------------
__global__ void k_node_gemm(const float* __restrict__ xin, const _Float16* __restrict__ w,
                            float* __restrict__ yout) {
  __shared__ __align__(16) _Float16 sX[WAVES][1024];
  const int lane = threadIdx.x & 31, wv = threadIdx.x >> 5;
  int tile = blockIdx.x * WAVES + wv;
  const int tmax = NN / 16 - 1;
  if (tile > tmax) tile = tmax;
  const long nb = (long)tile * 16;
  _Float16* X = sX[wv];
#pragma unroll
  for (int r = 0; r < 16; ++r) {
    float2 v = *(const float2*)(xin + (nb + r) * 64 + 2 * lane);
    st_pk16(X + r * 64 + 2 * lane, v.x, v.y);
  }
  __syncthreads();
  stageOut(X, w, yout, nb, lane);
}

// ---------------------------------------------------------------------------
// Kernel B: per-edge MLP chain -> i1, atomic segment-sum into p1_out
// ---------------------------------------------------------------------------
__global__ void k_edge1(const int* __restrict__ idx_i, const int* __restrict__ idx_j,
                        const float* __restrict__ p, const float* __restrict__ basis,
                        const _Float16* __restrict__ piw0, const _Float16* __restrict__ piw1,
                        const _Float16* __restrict__ iiw0, const _Float16* __restrict__ iiw1,
                        float* __restrict__ out) {
  __shared__ __align__(16) _Float16 sX[WAVES][1024];
  __shared__ __align__(16) _Float16 sH[WAVES][1024];
  __shared__ __align__(16) float    sX2[WAVES][1024];
  __shared__ float sBas[WAVES][128];
  __shared__ int   sI[WAVES][16];
  __shared__ int   sJ[WAVES][16];
  const int lane = threadIdx.x & 31, wv = threadIdx.x >> 5;
  const long e0 = ((long)blockIdx.x * WAVES + wv) * 16;   // grid exact: EE%(16*WAVES)==0

  if (lane < 16) { sI[wv][lane] = idx_i[e0 + lane]; sJ[wv][lane] = idx_j[e0 + lane]; }
#pragma unroll
  for (int q = 0; q < 4; ++q) {                  // 16x8 basis tile, contiguous
    int t = lane * 4 + q;
    sBas[wv][t] = basis[e0 * 8 + t];
  }
  __syncthreads();

  // gather + add: X = f16(p[idx_i] + p[idx_j])
  _Float16* X = sX[wv];
#pragma unroll
  for (int r = 0; r < 16; ++r) {
    float2 a = *(const float2*)(p + (long)sI[wv][r] * 64 + 2 * lane);
    float2 b = *(const float2*)(p + (long)sJ[wv][r] * 64 + 2 * lane);
    st_pk16(X + r * 64 + 2 * lane, a.x + b.x, a.y + b.y);
  }
  __syncthreads();

  // H = tanh(X @ pi_w0)
  stage64(X, piw0, sH[wv], lane, true);
  __syncthreads();

  // X2[m][c] = sum_b (H @ pi_w1)[m][c*8+b] * basis[m][b]   (pi_w1: 64x512)
  const int m = lane & 15, mb = (lane >> 4) * 8, hi8 = mb;
  v16h h0 = ldsA(sH[wv] + m * 64, 0, hi8);
  v16h h1 = ldsA(sH[wv] + m * 64, 32, hi8);
  float basR[8];
#pragma unroll
  for (int r = 0; r < 8; ++r) basR[r] = sBas[wv][(mb + r) * 8 + (lane & 7)];
  const int g = (lane >> 3) & 1;
  for (int nt = 0; nt < 32; ++nt) {
    v8f c = {};
    c = wmma16(h0, ldB(piw1, nt, lane), c);
    c = wmma16(h1, ldB(piw1, 32 + nt, lane), c);
#pragma unroll
    for (int r = 0; r < 8; ++r) {
      float t = c[r] * basR[r];
      t += __shfl_xor(t, 1);
      t += __shfl_xor(t, 2);
      t += __shfl_xor(t, 4);
      if ((lane & 7) == 0) sX2[wv][(mb + r) * 64 + 2 * nt + g] = t;
    }
  }
  __syncthreads();

  // X = f16(X2)  (packed converts, 2 elems/lane/iter)
#pragma unroll
  for (int q = 0; q < 16; ++q) {
    int t = q * 64 + 2 * lane;
    float2 v = *(const float2*)(&sX2[wv][t]);
    st_pk16(X + t, v.x, v.y);
  }
  __syncthreads();

  // H = tanh(X @ ii_w0)
  stage64(X, iiw0, sH[wv], lane, true);
  __syncthreads();

  // i1 = H @ ii_w1 ; write + atomic segment-sum into p1_out
  v16h g0 = ldsA(sH[wv] + m * 64, 0, hi8);
  v16h g1 = ldsA(sH[wv] + m * 64, 32, hi8);
#pragma unroll
  for (int nt = 0; nt < 4; ++nt) {
    v8f c = {};
    c = wmma16(g0, ldB(iiw1, nt, lane), c);
    c = wmma16(g1, ldB(iiw1, 4 + nt, lane), c);
    const int n = nt * 16 + m;
#pragma unroll
    for (int r = 0; r < 8; ++r) {
      float v = c[r];
      const long e = e0 + mb + r;
      out[O_I1 + e * 64 + n] = v;
      atomicAdd(out + O_P1 + (long)sI[wv][mb + r] * 64 + n, v);
    }
  }
}

// ---------------------------------------------------------------------------
// Kernel D: ix = px[i]@e_wi + px[j]@e_wj ; i3 = (diff * ix) @ e_ii
//           (= diff[e][d] * (ix @ e_ii)) ; atomic segment-sum into p3
// ---------------------------------------------------------------------------
__global__ void k_edge2(const int* __restrict__ idx_i, const int* __restrict__ idx_j,
                        const float* __restrict__ px, const float* __restrict__ diff,
                        const _Float16* __restrict__ wi, const _Float16* __restrict__ wj,
                        const _Float16* __restrict__ wii, float* __restrict__ out) {
  __shared__ __align__(16) _Float16 sXi[WAVES][1024];
  __shared__ __align__(16) _Float16 sXj[WAVES][1024];
  __shared__ __align__(16) _Float16 sY[WAVES][1024];
  __shared__ float sD[WAVES][48];
  __shared__ int   sI[WAVES][16];
  __shared__ int   sJ[WAVES][16];
  const int lane = threadIdx.x & 31, wv = threadIdx.x >> 5;
  const long e0 = ((long)blockIdx.x * WAVES + wv) * 16;

  if (lane < 16) {
    sI[wv][lane] = idx_i[e0 + lane];
    sJ[wv][lane] = idx_j[e0 + lane];
#pragma unroll
    for (int d = 0; d < 3; ++d) sD[wv][lane * 3 + d] = diff[e0 * 3 + lane * 3 + d];
  }
  __syncthreads();

#pragma unroll
  for (int r = 0; r < 16; ++r) {
    float2 a = *(const float2*)(px + (long)sI[wv][r] * 64 + 2 * lane);
    float2 b = *(const float2*)(px + (long)sJ[wv][r] * 64 + 2 * lane);
    st_pk16(sXi[wv] + r * 64 + 2 * lane, a.x, a.y);
    st_pk16(sXj[wv] + r * 64 + 2 * lane, b.x, b.y);
  }
  __syncthreads();

  // ix = Xi@e_wi + Xj@e_wj  (WMMA accumulate chain)
  const int m = lane & 15, mb = (lane >> 4) * 8, hi8 = mb;
  v16h ai0 = ldsA(sXi[wv] + m * 64, 0, hi8), ai1 = ldsA(sXi[wv] + m * 64, 32, hi8);
  v16h aj0 = ldsA(sXj[wv] + m * 64, 0, hi8), aj1 = ldsA(sXj[wv] + m * 64, 32, hi8);
#pragma unroll
  for (int nt = 0; nt < 4; ++nt) {
    v8f c = {};
    c = wmma16(ai0, ldB(wi, nt, lane), c);
    c = wmma16(ai1, ldB(wi, 4 + nt, lane), c);
    c = wmma16(aj0, ldB(wj, nt, lane), c);
    c = wmma16(aj1, ldB(wj, 4 + nt, lane), c);
    const int n = nt * 16 + m;
#pragma unroll
    for (int r = 0; r < 8; ++r) sY[wv][(mb + r) * 64 + n] = (_Float16)c[r];
  }
  __syncthreads();

  // y = ix @ e_ii ; i3[e][d][:] = diff[e][d] * y[e][:]
  v16h y0 = ldsA(sY[wv] + m * 64, 0, hi8), y1 = ldsA(sY[wv] + m * 64, 32, hi8);
#pragma unroll
  for (int nt = 0; nt < 4; ++nt) {
    v8f c = {};
    c = wmma16(y0, ldB(wii, nt, lane), c);
    c = wmma16(y1, ldB(wii, 4 + nt, lane), c);
    const int n = nt * 16 + m;
#pragma unroll
    for (int r = 0; r < 8; ++r) {
      const int row = mb + r;
      const long e = e0 + row;
      const long gi = sI[wv][row];
#pragma unroll
      for (int d = 0; d < 3; ++d) {
        float w = c[r] * sD[wv][row * 3 + d];
        out[O_I3 + (e * 3 + d) * 64 + n] = w;
        atomicAdd(out + O_P3 + (gi * 3 + d) * 64 + n, w);
      }
    }
  }
}

// ---------------------------------------------------------------------------
// Launch
// ---------------------------------------------------------------------------
extern "C" void kernel_launch(void* const* d_in, const int* in_sizes, int n_in,
                              void* d_out, int out_size, void* d_ws, size_t ws_size,
                              hipStream_t stream) {
  const int*   idx_i = (const int*)d_in[0];
  const int*   idx_j = (const int*)d_in[1];
  const float* p1    = (const float*)d_in[2];
  const float* basis = (const float*)d_in[3];
  const float* diff  = (const float*)d_in[4];
  const float* pp_w0 = (const float*)d_in[5];
  const float* pp_w1 = (const float*)d_in[6];
  const float* pi_w0 = (const float*)d_in[7];
  const float* pi_w1 = (const float*)d_in[8];
  const float* ii_w0 = (const float*)d_in[9];
  const float* ii_w1 = (const float*)d_in[10];
  const float* e_pp  = (const float*)d_in[11];
  const float* e_wi  = (const float*)d_in[12];
  const float* e_wj  = (const float*)d_in[13];
  const float* e_ii  = (const float*)d_in[14];

  float* out = (float*)d_out;
  char*  ws  = (char*)d_ws;
  float*     wsP  = (float*)(ws + WS_P_OFF);
  float*     wsPX = (float*)(ws + WS_PX_OFF);
  _Float16*  wsW  = (_Float16*)(ws + WS_W_OFF);

  // 1) swizzle weights to f16 B-fragment layout (recomputed every call)
  k_swz<<<16, 256, 0, stream>>>(pp_w0, wsW + W_PPW0, 64, 64);
  k_swz<<<16, 256, 0, stream>>>(pp_w1, wsW + W_PPW1, 64, 64);
  k_swz<<<16, 256, 0, stream>>>(pi_w0, wsW + W_PIW0, 64, 64);
  k_swz<<<128, 256, 0, stream>>>(pi_w1, wsW + W_PIW1, 64, 512);
  k_swz<<<16, 256, 0, stream>>>(ii_w0, wsW + W_IIW0, 64, 64);
  k_swz<<<16, 256, 0, stream>>>(ii_w1, wsW + W_IIW1, 64, 64);
  k_swz<<<16, 256, 0, stream>>>(e_pp,  wsW + W_EPP,  64, 64);
  k_swz<<<16, 256, 0, stream>>>(e_wi,  wsW + W_EWI,  64, 64);
  k_swz<<<16, 256, 0, stream>>>(e_wj,  wsW + W_EWJ,  64, 64);
  k_swz<<<16, 256, 0, stream>>>(e_ii,  wsW + W_EII,  64, 64);

  // 2) zero segment-sum accumulators in d_out
  k_zero<<<(NN * 256 + 255) / 256, 256, 0, stream>>>(out);

  // 3) node MLP: p
  const int nodeTiles = NN / 16;                        // 625
  const int nodeBlocks = (nodeTiles + WAVES - 1) / WAVES;
  k_node_mlp<<<nodeBlocks, 32 * WAVES, 0, stream>>>(p1, wsW + W_PPW0, wsW + W_PPW1, wsP);

  // 4) edge MLP chain -> i1, p1_out
  const int edgeBlocks = EE / 16 / WAVES;               // 5000, exact
  k_edge1<<<edgeBlocks, 32 * WAVES, 0, stream>>>(idx_i, idx_j, wsP, basis,
                                                 wsW + W_PIW0, wsW + W_PIW1,
                                                 wsW + W_IIW0, wsW + W_IIW1, out);

  // 5) px = p1_out @ e_pp
  k_node_gemm<<<nodeBlocks, 32 * WAVES, 0, stream>>>(out + O_P1, wsW + W_EPP, wsPX);

  // 6) edge interaction -> i3, p3
  k_edge2<<<edgeBlocks, 32 * WAVES, 0, stream>>>(idx_i, idx_j, wsPX, diff,
                                                 wsW + W_EWI, wsW + W_EWJ,
                                                 wsW + W_EII, out);
}
